// DiffNetpp_Encoder_57303453663961
// MI455X (gfx1250) — matrix operand order repaired and checked by reading, not compiled, not run on hardware
//
#include <hip/hip_runtime.h>
#include <hip/hip_bf16.h>
#include <math.h>

#define NU     50000
#define NI     50000
#define DD     64
#define NNZR   500000
#define NNZS   250000
#define NTOT   (NU + NI)
#define NLAY   2

typedef __attribute__((ext_vector_type(16))) __bf16 v16bf;
typedef __attribute__((ext_vector_type(8)))  float  v8f;

// ---------- helpers ----------
__device__ inline unsigned f32_ord(float f) {
    unsigned u = __float_as_uint(f);
    return (u & 0x80000000u) ? ~u : (u | 0x80000000u);
}
__device__ inline float ord_f32(unsigned u) {
    return (u & 0x80000000u) ? __uint_as_float(u & 0x7fffffffu)
                             : __uint_as_float(~u);
}

// ---------- small utility kernels ----------
__global__ void k_fill_f32(float* __restrict__ p, float v, int n) {
    for (int i = blockIdx.x * blockDim.x + threadIdx.x; i < n;
         i += gridDim.x * blockDim.x) p[i] = v;
}
__global__ void k_fill_u32(unsigned* __restrict__ p, unsigned v, int n) {
    for (int i = blockIdx.x * blockDim.x + threadIdx.x; i < n;
         i += gridDim.x * blockDim.x) p[i] = v;
}

// Pre-pack W1 [128x64] fp32 into bf16 B-fragment order:
// wf[((ks*4+nt)*32 + lane)*16 + e] = W1[(ks*32 + 16*h + e)*64 + nt*16 + n16]
// with n16 = lane&15, h = lane>>4.  Each lane then owns one contiguous v16bf.
__global__ void k_pack_w1_frags(const float* __restrict__ w1, __bf16* __restrict__ wf) {
    int i = blockIdx.x * blockDim.x + threadIdx.x;
    if (i >= 8192) return;
    int e    = i & 15;
    int lane = (i >> 4) & 31;
    int fr   = i >> 9;          // 0..15 = ks*4+nt
    int ks = fr >> 2, nt = fr & 3;
    int n16 = lane & 15, h = lane >> 4;
    int k = ks * 32 + 16 * h + e;
    int c = nt * 16 + n16;
    wf[i] = (__bf16)w1[k * 64 + c];
}

// ---------- WMMA MLP scorer: out[row] = relu(concat(A,B) @ W1 + b1) @ W2 + b2 ----------
// 128 threads = 4 waves; 16 rows per wave, 64 rows per block.
// Gather converts fp32 -> bf16 and stores directly in the WMMA A-fragment
// layout in LDS (pairs packed as b32).  A frags: 2x ds_load_b128 each.
// B frags come from the pre-packed global buffer: 2x global_load_b128 each.
__global__ __launch_bounds__(128) void k_mlp_score(
    const float* __restrict__ embA, const int* __restrict__ idxA,
    const float* __restrict__ embB, const int* __restrict__ idxB,
    const __bf16* __restrict__ wfrag, const float* __restrict__ b1,
    const float* __restrict__ w2,     const float* __restrict__ b2,
    float* __restrict__ out, int nE)
{
    __shared__ __bf16 ldsA[64 * 128];   // A fragments, 16 KB

    const int t    = threadIdx.x;
    const int base = blockIdx.x * 64;

    // gather: 64 rows x 64 feature-pairs; store packed bf16 pairs in frag layout
    for (int i = t; i < 64 * 64; i += 128) {
        int rl = i >> 6;             // local row 0..63
        int f  = (i & 63) * 2;       // even feature 0..126
        int e  = base + rl;
        float x0 = 0.f, x1 = 0.f;
        if (e < nE) {
            if (f < 64) {
                int r = idxA ? idxA[e] : e;
                x0 = embA[r * 64 + f];
                x1 = embA[r * 64 + f + 1];
            } else {
                int r = idxB ? idxB[e] : e;
                x0 = embB[r * 64 + (f - 64)];
                x1 = embB[r * 64 + (f - 63)];
            }
        }
        // A-layout inverse map (contiguous k = 16*h + el, same order as the W1 pack):
        int ks = f >> 5, kl = f & 31;
        int h  = kl >> 4;
        int el = kl & 15;                       // even; el+1 holds f+1
        int wv = rl >> 4, m = rl & 15;
        int lane = m + 16 * h;
        int dst = ((wv * 4 + ks) * 32 + lane) * 16 + el;   // even index -> 4B aligned
        union { __bf16 b[2]; unsigned u; } pk;
        pk.b[0] = (__bf16)x0;
        pk.b[1] = (__bf16)x1;
        *(unsigned*)&ldsA[dst] = pk.u;
    }
    __syncthreads();

    const int wave = t >> 5, lane = t & 31;
    const int n16  = lane & 15;          // B,D column n
    const int h    = lane >> 4;          // lane half

    // A fragments: contiguous vector loads from LDS
    v16bf afrag[4];
#pragma unroll
    for (int ks = 0; ks < 4; ++ks)
        afrag[ks] = *(const v16bf*)&ldsA[((wave * 4 + ks) * 32 + lane) * 16];

    // B fragments: contiguous vector loads from pre-packed global buffer
    const v16bf* wf = (const v16bf*)wfrag;
    v16bf bfrag[4][4];
#pragma unroll
    for (int ks = 0; ks < 4; ++ks)
#pragma unroll
        for (int nt = 0; nt < 4; ++nt)
            bfrag[ks][nt] = wf[(ks * 4 + nt) * 32 + lane];

    float part[8];
#pragma unroll
    for (int r = 0; r < 8; ++r) part[r] = 0.f;

#pragma unroll
    for (int nt = 0; nt < 4; ++nt) {
        v8f acc;
        float* cp = (float*)&acc;
        float c0 = b1[nt * 16 + n16];               // bias folded into C
#pragma unroll
        for (int r = 0; r < 8; ++r) cp[r] = c0;
#pragma unroll
        for (int ks = 0; ks < 4; ++ks)
            acc = __builtin_amdgcn_wmma_f32_16x16x32_bf16(
                false, afrag[ks], false, bfrag[ks][nt],
                (short)0, acc, false, false);
        float w2v = w2[nt * 16 + n16];
#pragma unroll
        for (int r = 0; r < 8; ++r) part[r] += fmaxf(cp[r], 0.f) * w2v;
    }
    // reduce across the 16 lanes of each half (D: VGPR r -> M = r + 8h, N = lane&15)
#pragma unroll
    for (int m = 1; m < 16; m <<= 1)
#pragma unroll
        for (int r = 0; r < 8; ++r) part[r] += __shfl_xor(part[r], m, 32);
    if (n16 == 0) {
        float b2v = b2[0];
#pragma unroll
        for (int r = 0; r < 8; ++r) {
            int e = base + wave * 16 + (r + 8 * h);
            if (e < nE) out[e] = part[r] + b2v;
        }
    }
}

// ---------- segment (row-wise) softmax over COO values ----------
__global__ void k_seg_max(const float* __restrict__ sc, const int* __restrict__ rows,
                          int roff, unsigned* __restrict__ rmax, int n) {
    int i = blockIdx.x * blockDim.x + threadIdx.x;
    if (i < n) atomicMax(&rmax[rows[i] + roff], f32_ord(sc[i]));
}
__global__ void k_seg_expsum(const float* __restrict__ sc, const int* __restrict__ rows,
                             int roff, const unsigned* __restrict__ rmax,
                             float* __restrict__ ebuf, float* __restrict__ rsum, int n) {
    int i = blockIdx.x * blockDim.x + threadIdx.x;
    if (i < n) {
        int r = rows[i] + roff;
        float e = expf(sc[i] - ord_f32(rmax[r]));
        ebuf[i] = e;
        atomicAdd(&rsum[r], e);
    }
}
__global__ void k_seg_norm(const float* __restrict__ basev, const float* __restrict__ ebuf,
                           const int* __restrict__ rows, int roff,
                           const float* __restrict__ rsum, float* __restrict__ outv, int n) {
    int i = blockIdx.x * blockDim.x + threadIdx.x;
    if (i < n) outv[i] = basev[i] * ebuf[i] / rsum[rows[i] + roff];
}

// ---------- global softmax (gates) ----------
__global__ __launch_bounds__(256) void k_red_max(const float* __restrict__ v, int n,
                                                 unsigned* __restrict__ slot) {
    float m = -3.4e38f;
    for (int i = blockIdx.x * 256 + threadIdx.x; i < n; i += gridDim.x * 256)
        m = fmaxf(m, v[i]);
#pragma unroll
    for (int s = 1; s < 32; s <<= 1) m = fmaxf(m, __shfl_xor(m, s, 32));
    __shared__ float wm[8];
    if ((threadIdx.x & 31) == 0) wm[threadIdx.x >> 5] = m;
    __syncthreads();
    if (threadIdx.x == 0) {
        for (int w = 1; w < 8; ++w) m = fmaxf(m, wm[w]);
        atomicMax(slot, f32_ord(m));
    }
}
__global__ __launch_bounds__(256) void k_exp_sumstore(const float* __restrict__ v, int n,
                                                      const unsigned* __restrict__ mslot,
                                                      float* __restrict__ eout,
                                                      float* __restrict__ sslot) {
    float mx = ord_f32(*mslot);
    float s = 0.f;
    for (int i = blockIdx.x * 256 + threadIdx.x; i < n; i += gridDim.x * 256) {
        float e = expf(v[i] - mx);
        eout[i] = e;
        s += e;
    }
#pragma unroll
    for (int m = 1; m < 32; m <<= 1) s += __shfl_xor(s, m, 32);
    __shared__ float ws[8];
    if ((threadIdx.x & 31) == 0) ws[threadIdx.x >> 5] = s;
    __syncthreads();
    if (threadIdx.x == 0) {
        for (int w = 1; w < 8; ++w) s += ws[w];
        atomicAdd(sslot, s);
    }
}
__global__ void k_scale_inv(float* __restrict__ e, int n, const float* __restrict__ sslot) {
    int i = blockIdx.x * blockDim.x + threadIdx.x;
    if (i < n) e[i] *= (1.f / *sslot);
}

// ---------- scatter SpMM: out[rows+roff] += vals * X[cols+coff] ----------
__global__ __launch_bounds__(256) void k_spmm(
    const float* __restrict__ vals, const int* __restrict__ rows,
    const int* __restrict__ cols, int roff, int coff,
    const float* __restrict__ X, float* __restrict__ out, int nnz)
{
    long total = (long)nnz * 64;
    for (long i = (long)blockIdx.x * 256 + threadIdx.x; i < total;
         i += (long)gridDim.x * 256) {
        long e = i >> 6; int d = (int)(i & 63);
        int r = rows[e] + roff, c = cols[e] + coff;
        atomicAdd(&out[(long)r * 64 + d], vals[e] * X[(long)c * 64 + d]);
    }
}

// ---------- init / gamma / update ----------
__global__ void k_init_emb(const float* __restrict__ ue, const float* __restrict__ ve,
                           float* __restrict__ U, float* __restrict__ V,
                           float* __restrict__ out) {
    int i = blockIdx.x * blockDim.x + threadIdx.x;
    if (i >= NTOT * 64) return;
    int node = i >> 6, d = i & 63;
    if (node < NU) {
        float v = ue[node * 64 + d];
        U[node * 64 + d] = v;
        out[(long)node * 192 + d] = v;
    } else {
        int m = node - NU;
        float v = ve[m * 64 + d];
        V[m * 64 + d] = v;
        out[(long)NU * 192 + (long)m * 192 + d] = v;
    }
}
__global__ void k_build_egam(const float* __restrict__ U, const float* __restrict__ V,
                             const float* __restrict__ g2, float* __restrict__ Eg) {
    int i = blockIdx.x * blockDim.x + threadIdx.x;
    if (i >= NTOT * 64) return;
    int node = i >> 6, d = i & 63;
    Eg[i] = (node < NU) ? U[node * 64 + d] * g2[node] : V[(node - NU) * 64 + d];
}
__global__ void k_update(const float* __restrict__ Eout, const float* __restrict__ Us,
                         const float* __restrict__ g1, float* __restrict__ U,
                         float* __restrict__ V, float* __restrict__ out, int layer) {
    int i = blockIdx.x * blockDim.x + threadIdx.x;
    if (i >= NTOT * 64) return;
    int node = i >> 6, d = i & 63;
    if (node < NU) {
        float un = Eout[i] + Us[node * 64 + d] * g1[node];
        U[node * 64 + d] = un;
        out[(long)node * 192 + layer * 64 + d] = un;
    } else {
        int m = node - NU;
        float vn = Eout[i] + V[m * 64 + d];
        V[m * 64 + d] = vn;
        out[(long)NU * 192 + (long)m * 192 + layer * 64 + d] = vn;
    }
}

// ---------------------------------------------------------------------------
extern "C" void kernel_launch(void* const* d_in, const int* in_sizes, int n_in,
                              void* d_out, int out_size, void* d_ws, size_t ws_size,
                              hipStream_t stream) {
    const float* user_emb = (const float*)d_in[0];
    const float* item_emb = (const float*)d_in[1];
    const float* r_vals   = (const float*)d_in[2];
    const float* s_vals   = (const float*)d_in[3];
    const float* adj_vals = (const float*)d_in[4];
    const float* w1s[4] = { (const float*)d_in[5],  (const float*)d_in[9],
                            (const float*)d_in[13], (const float*)d_in[17] };
    const float* b1s[4] = { (const float*)d_in[6],  (const float*)d_in[10],
                            (const float*)d_in[14], (const float*)d_in[18] };
    const float* w2s[4] = { (const float*)d_in[7],  (const float*)d_in[11],
                            (const float*)d_in[15], (const float*)d_in[19] };
    const float* b2s[4] = { (const float*)d_in[8],  (const float*)d_in[12],
                            (const float*)d_in[16], (const float*)d_in[20] };
    const int* r_rows = (const int*)d_in[21];
    const int* r_cols = (const int*)d_in[22];
    const int* s_rows = (const int*)d_in[23];
    const int* s_cols = (const int*)d_in[24];
    float* out = (float*)d_out;

    // workspace carve-out
    char* p = (char*)d_ws;
    auto alloc = [&](size_t bytes) -> void* {
        void* q = (void*)p;
        p += (bytes + 255) & ~(size_t)255;
        return q;
    };
    __bf16* w1fr[4];
    for (int i = 0; i < 4; ++i) w1fr[i] = (__bf16*)alloc(128 * 64 * sizeof(__bf16));
    float*    u_w     = (float*)alloc((size_t)NNZR * 4);
    float*    v_w     = (float*)alloc((size_t)NNZR * 4);
    float*    s_w     = (float*)alloc((size_t)NNZS * 4);
    float*    e_adj   = (float*)alloc((size_t)2 * NNZR * 4);
    float*    e_s     = (float*)alloc((size_t)NNZS * 4);
    float*    a_hat   = (float*)alloc((size_t)2 * NNZR * 4);
    float*    s_new   = (float*)alloc((size_t)NNZS * 4);
    unsigned* rmax_a  = (unsigned*)alloc((size_t)NTOT * 4);
    float*    rsum_a  = (float*)alloc((size_t)NTOT * 4);
    unsigned* rmax_s  = (unsigned*)alloc((size_t)NU * 4);
    float*    rsum_s  = (float*)alloc((size_t)NU * 4);
    float*    U_cur   = (float*)alloc((size_t)NU * 64 * 4);
    float*    V_cur   = (float*)alloc((size_t)NI * 64 * 4);
    float*    U_s     = (float*)alloc((size_t)NU * 64 * 4);
    float*    U_r     = (float*)alloc((size_t)NU * 64 * 4);
    float*    g_tmp   = (float*)alloc((size_t)NU * 4);
    float*    g1      = (float*)alloc((size_t)NU * 4);
    float*    g2      = (float*)alloc((size_t)NU * 4);
    float*    Egam    = (float*)alloc((size_t)NTOT * 64 * 4);
    float*    Eout    = (float*)alloc((size_t)NTOT * 64 * 4);
    unsigned* red_max = (unsigned*)alloc(256);
    float*    red_sum = (float*)alloc(256);

    auto nb = [](long total, int bs) { return (int)((total + bs - 1) / bs); };

    // 0. pack W1s into bf16 fragment order
    for (int i = 0; i < 4; ++i)
        k_pack_w1_frags<<<nb(8192, 256), 256, 0, stream>>>(w1s[i], w1fr[i]);

    // 1. edge scores (WMMA)
    k_mlp_score<<<nb(NNZR, 64), 128, 0, stream>>>(item_emb, r_cols, user_emb, r_rows,
        w1fr[0], b1s[0], w2s[0], b2s[0], v_w, NNZR);                 // mlp1
    k_mlp_score<<<nb(NNZR, 64), 128, 0, stream>>>(user_emb, r_rows, item_emb, r_cols,
        w1fr[2], b1s[2], w2s[2], b2s[2], u_w, NNZR);                 // mlp3
    k_mlp_score<<<nb(NNZS, 64), 128, 0, stream>>>(user_emb, s_rows, user_emb, s_cols,
        w1fr[1], b1s[1], w2s[1], b2s[1], s_w, NNZS);                 // mlp2

    // 2. S row-softmax -> s_new = s_vals .* softmax
    k_fill_u32<<<nb(NU, 256), 256, 0, stream>>>(rmax_s, 0u, NU);
    k_fill_f32<<<nb(NU, 256), 256, 0, stream>>>(rsum_s, 0.f, NU);
    k_seg_max<<<nb(NNZS, 256), 256, 0, stream>>>(s_w, s_rows, 0, rmax_s, NNZS);
    k_seg_expsum<<<nb(NNZS, 256), 256, 0, stream>>>(s_w, s_rows, 0, rmax_s, e_s, rsum_s, NNZS);
    k_seg_norm<<<nb(NNZS, 256), 256, 0, stream>>>(s_vals, e_s, s_rows, 0, rsum_s, s_new, NNZS);

    // 3. adj row-softmax (rows = [r_rows, NU + r_cols], scores = [u_w, v_w])
    k_fill_u32<<<nb(NTOT, 256), 256, 0, stream>>>(rmax_a, 0u, NTOT);
    k_fill_f32<<<nb(NTOT, 256), 256, 0, stream>>>(rsum_a, 0.f, NTOT);
    k_seg_max<<<nb(NNZR, 256), 256, 0, stream>>>(u_w, r_rows, 0, rmax_a, NNZR);
    k_seg_max<<<nb(NNZR, 256), 256, 0, stream>>>(v_w, r_cols, NU, rmax_a, NNZR);
    k_seg_expsum<<<nb(NNZR, 256), 256, 0, stream>>>(u_w, r_rows, 0, rmax_a, e_adj, rsum_a, NNZR);
    k_seg_expsum<<<nb(NNZR, 256), 256, 0, stream>>>(v_w, r_cols, NU, rmax_a, e_adj + NNZR, rsum_a, NNZR);
    k_seg_norm<<<nb(NNZR, 256), 256, 0, stream>>>(adj_vals, e_adj, r_rows, 0, rsum_a, a_hat, NNZR);
    k_seg_norm<<<nb(NNZR, 256), 256, 0, stream>>>(adj_vals + NNZR, e_adj + NNZR, r_cols, NU, rsum_a, a_hat + NNZR, NNZR);

    // 4. init embeddings + layer-0 output slice
    k_init_emb<<<nb((long)NTOT * 64, 256), 256, 0, stream>>>(user_emb, item_emb, U_cur, V_cur, out);

    // 5. propagation layers (n_layers fixed = 2 in reference setup)
    for (int l = 1; l <= NLAY; ++l) {
        k_fill_f32<<<nb((long)NU * 64, 256), 256, 0, stream>>>(U_s, 0.f, NU * 64);
        k_spmm<<<nb((long)NNZS * 64, 256), 256, 0, stream>>>(s_new, s_rows, s_cols, 0, 0, U_cur, U_s, NNZS);
        k_fill_f32<<<nb((long)NU * 64, 256), 256, 0, stream>>>(U_r, 0.f, NU * 64);
        k_spmm<<<nb((long)NNZR * 64, 256), 256, 0, stream>>>(r_vals, r_rows, r_cols, 0, 0, V_cur, U_r, NNZR);

        // g1 = softmax(mlp4(concat(U, U_s)))
        k_mlp_score<<<nb(NU, 64), 128, 0, stream>>>(U_cur, nullptr, U_s, nullptr,
            w1fr[3], b1s[3], w2s[3], b2s[3], g_tmp, NU);
        k_fill_u32<<<1, 32, 0, stream>>>(red_max, 0u, 1);
        k_fill_f32<<<1, 32, 0, stream>>>(red_sum, 0.f, 1);
        k_red_max<<<256, 256, 0, stream>>>(g_tmp, NU, red_max);
        k_exp_sumstore<<<256, 256, 0, stream>>>(g_tmp, NU, red_max, g1, red_sum);
        k_scale_inv<<<nb(NU, 256), 256, 0, stream>>>(g1, NU, red_sum);

        // g2 = softmax(mlp4(concat(U, U_r)))
        k_mlp_score<<<nb(NU, 64), 128, 0, stream>>>(U_cur, nullptr, U_r, nullptr,
            w1fr[3], b1s[3], w2s[3], b2s[3], g_tmp, NU);
        k_fill_u32<<<1, 32, 0, stream>>>(red_max, 0u, 1);
        k_fill_f32<<<1, 32, 0, stream>>>(red_sum, 0.f, 1);
        k_red_max<<<256, 256, 0, stream>>>(g_tmp, NU, red_max);
        k_exp_sumstore<<<256, 256, 0, stream>>>(g_tmp, NU, red_max, g2, red_sum);
        k_scale_inv<<<nb(NU, 256), 256, 0, stream>>>(g2, NU, red_sum);

        // E = concat(U*g2, V); Eout = A_hat @ E
        k_build_egam<<<nb((long)NTOT * 64, 256), 256, 0, stream>>>(U_cur, V_cur, g2, Egam);
        k_fill_f32<<<nb((long)NTOT * 64, 256), 256, 0, stream>>>(Eout, 0.f, NTOT * 64);
        k_spmm<<<nb((long)NNZR * 64, 256), 256, 0, stream>>>(a_hat, r_rows, r_cols, 0, NU, Egam, Eout, NNZR);
        k_spmm<<<nb((long)NNZR * 64, 256), 256, 0, stream>>>(a_hat + NNZR, r_cols, r_rows, NU, 0, Egam, Eout, NNZR);

        // U = Eout_u + U_s*g1 ; V = Eout_v + V ; write layer slice
        k_update<<<nb((long)NTOT * 64, 256), 256, 0, stream>>>(Eout, U_s, g1, U_cur, V_cur, out, l);
    }
    (void)in_sizes; (void)n_in; (void)out_size; (void)ws_size;
}
